// TCNet_6777458393642
// MI455X (gfx1250) — matrix operand order repaired and checked by reading, not compiled, not run on hardware
//
#include <hip/hip_runtime.h>
#include <hip/hip_bf16.h>

// TCNet Tucker-fusion pipeline for MI455X (gfx1250, wave32, WMMA).
// All heavy contractions run on v_wmma_f32_16x16x32_bf16 (bf16 in, fp32 acc),
// with 2x2 register blocking (one wave -> 32x32 output macro-tile).
// Dims: B=16, N=64, D=H=512, R=8, HK=64, G=2.

typedef __bf16 bf16;
typedef __attribute__((ext_vector_type(16))) __bf16 v16bf;
typedef __attribute__((ext_vector_type(8)))  float  v8f;

// ---------------- WMMA fragment helpers (wave32, 16x16x32 bf16) -------------
// A (16x32, MxK): lane m (0..15) holds K = {kh..kh+15},
// kh = 16*(lane>=16).  Storage: A[m*ld + k*es].
static __device__ __forceinline__ v16bf load_a_frag(const bf16* __restrict__ base,
                                                    int ld, int es) {
  int lane = threadIdx.x & 31;
  int m = lane & 15;
  int kh = (lane >> 4) << 4;             // 0 or 16
  const bf16* p = base + m * ld;
  v16bf o;
#pragma unroll
  for (int i = 0; i < 16; ++i) o[i]     = p[(kh + i) * es];
  return o;
}

// B (32x16, KxN) given as BT in memory: [N x K] row-major, lane n reads a
// contiguous run of 16 K-values (K 0..15 for lanes 0-15, 16..31 for 16-31).
static __device__ __forceinline__ v16bf load_b_nk(const bf16* __restrict__ base,
                                                  int ld) {
  int lane = threadIdx.x & 31;
  int n = lane & 15;
  int k0 = (lane >> 4) << 4;             // 0 or 16
  const bf16* p = base + n * ld + k0;
  v16bf o;
#pragma unroll
  for (int i = 0; i < 16; ++i) o[i] = p[i];
  return o;
}

// B given naturally as [K x N] row-major: lane n reads stride-ld K-values.
static __device__ __forceinline__ v16bf load_b_kn(const bf16* __restrict__ base,
                                                  int ld) {
  int lane = threadIdx.x & 31;
  int n = lane & 15;
  int k0 = (lane >> 4) << 4;
  const bf16* p = base + (size_t)k0 * ld + n;
  v16bf o;
#pragma unroll
  for (int i = 0; i < 16; ++i) o[i] = p[(size_t)i * ld];
  return o;
}

static __device__ __forceinline__ v8f wmma_bf16(v16bf a, v16bf b, v8f c) {
  return __builtin_amdgcn_wmma_f32_16x16x32_bf16(false, a, false, b,
                                                 (short)0, c, false, false);
}

// C/D (16x16 f32): lane: n = lane&15, m = 8*(lane>=16) + vgpr_index.

// ---------------- small utility kernels -------------------------------------

__global__ void k_cvt(const float* __restrict__ src, bf16* __restrict__ dst, int n) {
  int i = blockIdx.x * blockDim.x + threadIdx.x;
  if (i < n) dst[i] = (bf16)src[i];
}

// T_g fp32 [R][H=64][K*L*G=8192]  ->  Tt bf16 [R][8192][64]  (transposed)
__global__ void k_cvt_T(const float* __restrict__ T, bf16* __restrict__ Tt) {
  int i = blockIdx.x * blockDim.x + threadIdx.x;      // 8*64*8192 = 4194304
  if (i >= 8 * 64 * 8192) return;
  int n = i & 8191;
  int h = (i >> 13) & 63;
  int r = i >> 19;
  Tt[((size_t)r * 8192 + n) * 64 + h] = (bf16)T[i];
}

// weight-norm scales: scales[0..2] = g/||V_{v,q,a}||, [3..10]=gn_v[r]/||Vn_v[r]||,
// [11..18]=gn_qa[r]/||Vn_qa[r]||
__global__ void k_norms(const float* __restrict__ Vv, const float* __restrict__ gv,
                        const float* __restrict__ Vq, const float* __restrict__ gq,
                        const float* __restrict__ Va, const float* __restrict__ ga,
                        const float* __restrict__ Vnv, const float* __restrict__ gnv,
                        const float* __restrict__ Vnqa, const float* __restrict__ gnqa,
                        float* __restrict__ scales) {
  __shared__ float red[256];
  int bid = blockIdx.x;
  const float* src; int n; float g;
  if (bid == 0)      { src = Vv; n = 512 * 512; g = gv[0]; }
  else if (bid == 1) { src = Vq; n = 512 * 512; g = gq[0]; }
  else if (bid == 2) { src = Va; n = 512 * 512; g = ga[0]; }
  else if (bid < 11) { int r = bid - 3;  src = Vnv  + r * 64 * 512; n = 64 * 512; g = gnv[r]; }
  else               { int r = bid - 11; src = Vnqa + r * 64 * 512; n = 64 * 512; g = gnqa[r]; }
  float s = 0.f;
  for (int i = threadIdx.x; i < n; i += 256) { float x = src[i]; s += x * x; }
  red[threadIdx.x] = s;
  __syncthreads();
  for (int st = 128; st > 0; st >>= 1) {
    if (threadIdx.x < st) red[threadIdx.x] += red[threadIdx.x + st];
    __syncthreads();
  }
  if (threadIdx.x == 0) scales[bid] = g / sqrtf(red[0]);
}

// ---------------- generic GEMM: out = relu(scale * X*W^T + bias), bf16 out --
// X:[M x K] bf16 row-major, W:[N x K] bf16 row-major.
// One wave per 32x32 macro-tile (2x2 x 16x16 WMMA tiles).
__global__ void k_gemm_bias_relu(const bf16* __restrict__ X, const bf16* __restrict__ W,
                                 const float* __restrict__ bias,
                                 const float* __restrict__ scale_p,
                                 bf16* __restrict__ out, int N, int K) {
  int wave = threadIdx.x >> 5;
  int tile = blockIdx.x * (blockDim.x >> 5) + wave;
  int nt = N >> 5;
  int tm = tile / nt, tn = tile % nt;
  const bf16* A0 = X + (size_t)(tm * 32) * K;
  const bf16* A1 = A0 + (size_t)16 * K;
  const bf16* B0 = W + (size_t)(tn * 32) * K;
  const bf16* B1 = B0 + (size_t)16 * K;
  v8f acc00 = {}, acc01 = {}, acc10 = {}, acc11 = {};
  for (int k0 = 0; k0 < K; k0 += 32) {
    __builtin_prefetch(B0 + k0 + 64, 0, 3);
    __builtin_prefetch(B1 + k0 + 64, 0, 3);
    v16bf a0 = load_a_frag(A0 + k0, K, 1);
    v16bf a1 = load_a_frag(A1 + k0, K, 1);
    v16bf b0 = load_b_nk(B0 + k0, K);
    v16bf b1 = load_b_nk(B1 + k0, K);
    acc00 = wmma_bf16(a0, b0, acc00);
    acc01 = wmma_bf16(a0, b1, acc01);
    acc10 = wmma_bf16(a1, b0, acc10);
    acc11 = wmma_bf16(a1, b1, acc11);
  }
  float s = scale_p[0];
  int lane = threadIdx.x & 31;
  int n = lane & 15, m0 = (lane >> 4) << 3;
  v8f accs[2][2] = {{acc00, acc01}, {acc10, acc11}};
#pragma unroll
  for (int j = 0; j < 2; ++j) {
    float bv = bias[tn * 32 + j * 16 + n];
#pragma unroll
    for (int i = 0; i < 2; ++i) {
#pragma unroll
      for (int t = 0; t < 8; ++t) {
        float v = fmaxf(accs[i][j][t] * s + bv, 0.f);
        out[(size_t)(tm * 32 + i * 16 + m0 + t) * N + tn * 32 + j * 16 + n] = (bf16)v;
      }
    }
  }
}

// ---------------- stage 3 (Tucker core), chunked per batch element b --------
// S1: C1t[r][v][lg][k] = sum_h v_r[b,r,v,h] * T[r,h,(k,lg)]
__global__ void k_s1(const bf16* __restrict__ vr /*[R][1024][64]*/,
                     const bf16* __restrict__ Tt /*[R][8192][64]*/,
                     bf16* __restrict__ C1t /*[R][64][128][64]*/, int b) {
  int r = blockIdx.z;
  int wave = threadIdx.x >> 5;
  int tile = blockIdx.x * (blockDim.x >> 5) + wave;   // 2 (m) * 256 (n) macro
  int tm = tile >> 8, tn = tile & 255;
  const bf16* A0 = vr + ((size_t)r * 1024 + b * 64 + tm * 32) * 64;
  const bf16* A1 = A0 + 16 * 64;
  const bf16* B0 = Tt + ((size_t)r * 8192 + tn * 32) * 64;
  const bf16* B1 = B0 + 16 * 64;
  v8f acc00 = {}, acc01 = {}, acc10 = {}, acc11 = {};
  for (int k0 = 0; k0 < 64; k0 += 32) {
    __builtin_prefetch(B0 + k0 + 2048, 0, 3);
    v16bf a0 = load_a_frag(A0 + k0, 64, 1);
    v16bf a1 = load_a_frag(A1 + k0, 64, 1);
    v16bf b0 = load_b_nk(B0 + k0, 64);
    v16bf b1 = load_b_nk(B1 + k0, 64);
    acc00 = wmma_bf16(a0, b0, acc00);
    acc01 = wmma_bf16(a0, b1, acc01);
    acc10 = wmma_bf16(a1, b0, acc10);
    acc11 = wmma_bf16(a1, b1, acc11);
  }
  int lane = threadIdx.x & 31;
  int ncol = lane & 15, m0 = (lane >> 4) << 3;
  v8f accs[2][2] = {{acc00, acc01}, {acc10, acc11}};
#pragma unroll
  for (int j = 0; j < 2; ++j) {
    int n = tn * 32 + j * 16 + ncol;
    int kk = n >> 7, lg = n & 127;                    // n = kk*128 + lg
#pragma unroll
    for (int i = 0; i < 2; ++i) {
#pragma unroll
      for (int t = 0; t < 8; ++t)
        C1t[(((size_t)r * 64 + tm * 32 + i * 16 + m0 + t) * 128 + lg) * 64 + kk] =
            (bf16)accs[i][j][t];
    }
  }
}

// S2: C2[r][v*64+q][lg] = sum_k q_r[b,r,q,k] * C1t[r][v][lg][k]
__global__ void k_s2(const bf16* __restrict__ qr, const bf16* __restrict__ C1t,
                     bf16* __restrict__ C2 /*[R][4096][128]*/, int b) {
  int r = blockIdx.z, v = blockIdx.y;
  int wave = threadIdx.x >> 5;
  int tile = wave;                                    // 2 (q) * 4 (lg) = 8 macro
  int tm = tile >> 2, tn = tile & 3;
  const bf16* A0 = qr  + ((size_t)r * 1024 + b * 64 + tm * 32) * 64;
  const bf16* A1 = A0 + 16 * 64;
  const bf16* B0 = C1t + (((size_t)r * 64 + v) * 128 + tn * 32) * 64;
  const bf16* B1 = B0 + 16 * 64;
  v8f acc00 = {}, acc01 = {}, acc10 = {}, acc11 = {};
  for (int k0 = 0; k0 < 64; k0 += 32) {
    v16bf a0 = load_a_frag(A0 + k0, 64, 1);
    v16bf a1 = load_a_frag(A1 + k0, 64, 1);
    v16bf b0 = load_b_nk(B0 + k0, 64);
    v16bf b1 = load_b_nk(B1 + k0, 64);
    acc00 = wmma_bf16(a0, b0, acc00);
    acc01 = wmma_bf16(a0, b1, acc01);
    acc10 = wmma_bf16(a1, b0, acc10);
    acc11 = wmma_bf16(a1, b1, acc11);
  }
  int lane = threadIdx.x & 31;
  int ncol = lane & 15, m0 = (lane >> 4) << 3;
  v8f accs[2][2] = {{acc00, acc01}, {acc10, acc11}};
#pragma unroll
  for (int j = 0; j < 2; ++j)
#pragma unroll
    for (int i = 0; i < 2; ++i)
#pragma unroll
      for (int t = 0; t < 8; ++t)
        C2[((size_t)r * 4096 + v * 64 + tm * 32 + i * 16 + m0 + t) * 128 +
           tn * 32 + j * 16 + ncol] = (bf16)accs[i][j][t];
}

// S3: f_emb[b][vq][m][g] = sum_r sum_l C2[r][vq][l*2+g] * a_r[b,r,m,l]
__global__ void k_s3(const bf16* __restrict__ C2, const bf16* __restrict__ ar,
                     float* __restrict__ fout, int b) {
  int g = blockIdx.z;
  int wave = threadIdx.x >> 5;
  int tile = blockIdx.x * (blockDim.x >> 5) + wave;   // 128 (vq) * 2 (m) macro
  int tm = tile >> 1, tn = tile & 1;
  v8f acc00 = {}, acc01 = {}, acc10 = {}, acc11 = {};
  for (int r = 0; r < 8; ++r) {
    const bf16* A0 = C2 + ((size_t)r * 4096 + tm * 32) * 128 + g;   // es=2
    const bf16* A1 = A0 + 16 * 128;
    const bf16* B0 = ar + ((size_t)r * 1024 + b * 64 + tn * 32) * 64;
    const bf16* B1 = B0 + 16 * 64;
    for (int k0 = 0; k0 < 64; k0 += 32) {
      v16bf a0 = load_a_frag(A0 + k0 * 2, 128, 2);
      v16bf a1 = load_a_frag(A1 + k0 * 2, 128, 2);
      v16bf b0 = load_b_nk(B0 + k0, 64);
      v16bf b1 = load_b_nk(B1 + k0, 64);
      acc00 = wmma_bf16(a0, b0, acc00);
      acc01 = wmma_bf16(a0, b1, acc01);
      acc10 = wmma_bf16(a1, b0, acc10);
      acc11 = wmma_bf16(a1, b1, acc11);
    }
  }
  int lane = threadIdx.x & 31;
  int ncol = lane & 15, m0 = (lane >> 4) << 3;
  v8f accs[2][2] = {{acc00, acc01}, {acc10, acc11}};
#pragma unroll
  for (int j = 0; j < 2; ++j)
#pragma unroll
    for (int i = 0; i < 2; ++i)
#pragma unroll
      for (int t = 0; t < 8; ++t)
        fout[(((size_t)b * 4096 + tm * 32 + i * 16 + m0 + t) * 64 +
              tn * 32 + j * 16 + ncol) * 2 + g] = accs[i][j][t];
}

// ---------------- stage 4: logits ------------------------------------------
__global__ void k_sum_g(const float* __restrict__ f, bf16* __restrict__ F, int n) {
  int i = blockIdx.x * blockDim.x + threadIdx.x;
  if (i < n) F[i] = (bf16)(f[2 * i] + f[2 * i + 1]);
}

// G1[vq][d] = sum_a F[b][vq][a] * a_t[b,a,d]   (per b)
__global__ void k_g1(const bf16* __restrict__ F, const bf16* __restrict__ at,
                     float* __restrict__ G1, int b) {
  int wave = threadIdx.x >> 5;
  int tile = blockIdx.x * (blockDim.x >> 5) + wave;   // 128 (vq) * 16 (d) macro
  int tm = tile >> 4, tn = tile & 15;
  const bf16* A0 = F  + ((size_t)b * 4096 + tm * 32) * 64;
  const bf16* A1 = A0 + 16 * 64;
  const bf16* B0 = at + (size_t)b * 64 * 512 + tn * 32;   // [K=a x N=d] row-major
  const bf16* B1 = B0 + 16;
  v8f acc00 = {}, acc01 = {}, acc10 = {}, acc11 = {};
  for (int k0 = 0; k0 < 64; k0 += 32) {
    v16bf a0 = load_a_frag(A0 + k0, 64, 1);
    v16bf a1 = load_a_frag(A1 + k0, 64, 1);
    v16bf b0 = load_b_kn(B0 + (size_t)k0 * 512, 512);
    v16bf b1 = load_b_kn(B1 + (size_t)k0 * 512, 512);
    acc00 = wmma_bf16(a0, b0, acc00);
    acc01 = wmma_bf16(a0, b1, acc01);
    acc10 = wmma_bf16(a1, b0, acc10);
    acc11 = wmma_bf16(a1, b1, acc11);
  }
  int lane = threadIdx.x & 31;
  int n = lane & 15, m0 = (lane >> 4) << 3;
  v8f accs[2][2] = {{acc00, acc01}, {acc10, acc11}};
#pragma unroll
  for (int j = 0; j < 2; ++j)
#pragma unroll
    for (int i = 0; i < 2; ++i)
#pragma unroll
      for (int t = 0; t < 8; ++t)
        G1[(size_t)(tm * 32 + i * 16 + m0 + t) * 512 + tn * 32 + j * 16 + n] =
            accs[i][j][t];
}

// lraw[b,d] = sum_v v_t[b,v,d] * sum_q q_t[b,q,d] * G1[v*64+q][d]
__global__ void k_logits_b(const float* __restrict__ G1, const bf16* __restrict__ vt,
                           const bf16* __restrict__ qt, float* __restrict__ lraw, int b) {
  int d = blockIdx.x * blockDim.x + threadIdx.x;      // 512
  float acc = 0.f;
  for (int v = 0; v < 64; ++v) {
    float sq = 0.f;
    for (int q = 0; q < 64; ++q)
      sq += G1[(size_t)(v * 64 + q) * 512 + d] * (float)qt[(size_t)(b * 64 + q) * 512 + d];
    acc += (float)vt[(size_t)(b * 64 + v) * 512 + d] * sq;
  }
  lraw[b * 512 + d] = acc;
}

// training-mode BatchNorm over batch dim (16), eps=1e-5
__global__ void k_bn(const float* __restrict__ lraw, const float* __restrict__ gamma,
                     const float* __restrict__ beta, float* __restrict__ out) {
  int d = threadIdx.x;                                // 512
  float x[16], mu = 0.f;
#pragma unroll
  for (int b = 0; b < 16; ++b) { x[b] = lraw[b * 512 + d]; mu += x[b]; }
  mu *= (1.f / 16.f);
  float var = 0.f;
#pragma unroll
  for (int b = 0; b < 16; ++b) { float t = x[b] - mu; var += t * t; }
  var *= (1.f / 16.f);
  float sc = gamma[d] / sqrtf(var + 1e-5f);
  float bt = beta[d];
#pragma unroll
  for (int b = 0; b < 16; ++b)
    out[8388608 + b * 512 + d] = (x[b] - mu) * sc + bt;
}

// ---------------- driver -----------------------------------------------------
extern "C" void kernel_launch(void* const* d_in, const int* in_sizes, int n_in,
                              void* d_out, int out_size, void* d_ws, size_t ws_size,
                              hipStream_t stream) {
  (void)in_sizes; (void)n_in; (void)out_size; (void)ws_size;

  const float* v_in  = (const float*)d_in[0];
  const float* q_in  = (const float*)d_in[1];
  const float* a_in  = (const float*)d_in[2];
  const float* V_v   = (const float*)d_in[3];
  const float* g_v   = (const float*)d_in[4];
  const float* b_v   = (const float*)d_in[5];
  const float* V_q   = (const float*)d_in[6];
  const float* g_q   = (const float*)d_in[7];
  const float* b_q   = (const float*)d_in[8];
  const float* V_a   = (const float*)d_in[9];
  const float* g_a   = (const float*)d_in[10];
  const float* b_a   = (const float*)d_in[11];
  const float* Vn_v  = (const float*)d_in[12];
  const float* gn_v  = (const float*)d_in[13];
  const float* bn_v  = (const float*)d_in[14];
  const float* Vn_qa = (const float*)d_in[15];
  const float* gn_qa = (const float*)d_in[16];
  const float* bn_qa = (const float*)d_in[17];
  const float* T_g   = (const float*)d_in[18];
  const float* gamma = (const float*)d_in[19];
  const float* beta  = (const float*)d_in[20];
  float* out = (float*)d_out;

  // ---- carve workspace (~53 MB) ----
  char* w = (char*)d_ws;
  auto carve = [&](size_t sz) { char* p = w; w += (sz + 255) & ~(size_t)255; return p; };
  float* scales = (float*)carve(19 * 4);
  bf16* vbf   = (bf16*)carve((size_t)1024 * 512 * 2);
  bf16* qbf   = (bf16*)carve((size_t)1024 * 512 * 2);
  bf16* abf   = (bf16*)carve((size_t)1024 * 512 * 2);
  bf16* Wv    = (bf16*)carve((size_t)512 * 512 * 2);
  bf16* Wq    = (bf16*)carve((size_t)512 * 512 * 2);
  bf16* Wa    = (bf16*)carve((size_t)512 * 512 * 2);
  bf16* Wnv   = (bf16*)carve((size_t)8 * 64 * 512 * 2);
  bf16* Wnqa  = (bf16*)carve((size_t)8 * 64 * 512 * 2);
  bf16* Tt    = (bf16*)carve((size_t)8 * 8192 * 64 * 2);
  bf16* vt    = (bf16*)carve((size_t)1024 * 512 * 2);
  bf16* qt    = (bf16*)carve((size_t)1024 * 512 * 2);
  bf16* at    = (bf16*)carve((size_t)1024 * 512 * 2);
  bf16* vr    = (bf16*)carve((size_t)8 * 1024 * 64 * 2);
  bf16* qr    = (bf16*)carve((size_t)8 * 1024 * 64 * 2);
  bf16* ar    = (bf16*)carve((size_t)8 * 1024 * 64 * 2);
  bf16* C1t   = (bf16*)carve((size_t)8 * 64 * 128 * 64 * 2);
  bf16* C2    = (bf16*)carve((size_t)8 * 4096 * 128 * 2);
  bf16* Fsum  = (bf16*)carve((size_t)16 * 4096 * 64 * 2);
  float* G1   = (float*)carve((size_t)4096 * 512 * 4);
  float* lraw = (float*)carve((size_t)16 * 512 * 4);

  // ---- weight-norm scales + bf16 conversions ----
  k_norms<<<19, 256, 0, stream>>>(V_v, g_v, V_q, g_q, V_a, g_a,
                                  Vn_v, gn_v, Vn_qa, gn_qa, scales);
  k_cvt<<<2048, 256, 0, stream>>>(v_in, vbf, 524288);
  k_cvt<<<2048, 256, 0, stream>>>(q_in, qbf, 524288);
  k_cvt<<<2048, 256, 0, stream>>>(a_in, abf, 524288);
  k_cvt<<<1024, 256, 0, stream>>>(V_v, Wv, 262144);
  k_cvt<<<1024, 256, 0, stream>>>(V_q, Wq, 262144);
  k_cvt<<<1024, 256, 0, stream>>>(V_a, Wa, 262144);
  k_cvt<<<1024, 256, 0, stream>>>(Vn_v, Wnv, 262144);
  k_cvt<<<1024, 256, 0, stream>>>(Vn_qa, Wnqa, 262144);
  k_cvt_T<<<16384, 256, 0, stream>>>(T_g, Tt);

  // ---- stage 1: v_t/q_t/a_t = relu(s * X Vraw^T + b)  (M=1024,N=512,K=512)
  // macro-tiles: 32x16 = 512 -> 64 blocks x 8 waves
  k_gemm_bias_relu<<<64, 256, 0, stream>>>(vbf, Wv, b_v, scales + 0, vt, 512, 512);
  k_gemm_bias_relu<<<64, 256, 0, stream>>>(qbf, Wq, b_q, scales + 1, qt, 512, 512);
  k_gemm_bias_relu<<<64, 256, 0, stream>>>(abf, Wa, b_a, scales + 2, at, 512, 512);

  // ---- stage 2: per-rank projections (M=1024,N=64,K=512), 24 GEMMs
  // macro-tiles: 32x2 = 64 -> 8 blocks x 8 waves
  for (int r = 0; r < 8; ++r) {
    k_gemm_bias_relu<<<8, 256, 0, stream>>>(vt, Wnv + (size_t)r * 64 * 512,
        bn_v + r * 64, scales + 3 + r, vr + (size_t)r * 1024 * 64, 64, 512);
    k_gemm_bias_relu<<<8, 256, 0, stream>>>(qt, Wnqa + (size_t)r * 64 * 512,
        bn_qa + r * 64, scales + 11 + r, qr + (size_t)r * 1024 * 64, 64, 512);
    k_gemm_bias_relu<<<8, 256, 0, stream>>>(at, Wnqa + (size_t)r * 64 * 512,
        bn_qa + r * 64, scales + 11 + r, ar + (size_t)r * 1024 * 64, 64, 512);
  }

  // ---- stage 3: Tucker core, chunked per b (C1t/C2 reused each iteration)
  for (int b = 0; b < 16; ++b) {
    k_s1<<<dim3(64, 1, 8), 256, 0, stream>>>(vr, Tt, C1t, b);    // 512 macro / r
    k_s2<<<dim3(1, 64, 8), 256, 0, stream>>>(qr, C1t, C2, b);    // 8 macro / (v,r)
    k_s3<<<dim3(32, 1, 2), 256, 0, stream>>>(C2, ar, out, b);    // 256 macro / g
  }

  // ---- stage 4: logits
  k_sum_g<<<16384, 256, 0, stream>>>(out, Fsum, 16 * 4096 * 64);
  for (int b = 0; b < 16; ++b) {
    k_g1<<<256, 256, 0, stream>>>(Fsum, at, G1, b);              // 2048 macro
    k_logits_b<<<2, 256, 0, stream>>>(G1, vt, qt, lraw, b);
  }
  k_bn<<<1, 512, 0, stream>>>(lraw, gamma, beta, out);
}
